// LSTM_63436666962432
// MI455X (gfx1250) — compile-verified, matrix operand order checked by reading, not run on hardware
//
#include <hip/hip_runtime.h>

// ---------------------------------------------------------------------------
// LSTM for MI455X (gfx1250, wave32, WMMA).
//   Phase 1: xproj = x @ Wx + b as a bf16 WMMA GEMM (f32 accum), stored bf16
//            in (T, N, 4H) order so the scan reads contiguous rows per step.
//   Phase 2: persistent scan kernel, 64 single-wave WGs; each wave owns
//            (batch-half, 16 hidden units), caches its 64KB Wh^T slice in LDS,
//            keeps cell state in VGPRs, grid-syncs via an L2 atomic per step.
// ---------------------------------------------------------------------------

typedef __attribute__((ext_vector_type(16))) __bf16 v16bf;
typedef __attribute__((ext_vector_type(8)))  __bf16 v8bf;
typedef __attribute__((ext_vector_type(4)))  __bf16 v4bf;
typedef __attribute__((ext_vector_type(8)))  float  v8f;
typedef __attribute__((ext_vector_type(4)))  float  v4f;

constexpr int NBATCH = 32;
constexpr int TSTEPS = 2048;
constexpr int DDIM   = 512;
constexpr int HDIM   = 512;
constexpr int GDIM   = 2048;   // 4*H

// Workspace layout (bytes), all offsets 256-aligned. Total ~324 MiB.
constexpr size_t OFF_CNT = 0;                                  // 256 B
constexpr size_t OFF_HBF = 256;                                // 2 * 32*512*2 = 64 KiB
constexpr size_t OFF_WXT = OFF_HBF + 2ull * NBATCH * HDIM * 2; // 2 MiB
constexpr size_t OFF_WHT = OFF_WXT + (size_t)GDIM * DDIM * 2;  // 2 MiB
constexpr size_t OFF_XB  = OFF_WHT + (size_t)GDIM * DDIM * 2;  // 64 MiB
constexpr size_t OFF_XP  = OFF_XB  + (size_t)NBATCH * TSTEPS * DDIM * 2; // 256 MiB

// ---- fragment helpers (per CDNA5 ISA 16-bit A/B layouts, wave32) ----------
// Lane L (0..15): row/col = L, holds K {0..7, 16..23} in its 8 VGPRs.
// Lane L (16..31): row/col = L-16, holds K {8..15, 24..31}.
__device__ __forceinline__ v16bf load_frag(const __bf16* __restrict__ base,
                                           int row, int ld, int k0) {
  const int lane = threadIdx.x & 31;
  const int half = lane >> 4;
  const __bf16* p = base + (size_t)row * ld + k0 + half * 8;
  v8bf lo = *(const v8bf*)p;          // K = k0 + half*8 .. +7
  v8bf hi = *(const v8bf*)(p + 16);   // K = k0 + 16 + half*8 .. +7
  return __builtin_shufflevector(lo, hi, 0,1,2,3,4,5,6,7,8,9,10,11,12,13,14,15);
}

__device__ __forceinline__ v8f wmma_bf16(v16bf a, v16bf b, v8f c) {
  return __builtin_amdgcn_wmma_f32_16x16x32_bf16(
      /*neg_a=*/false, a, /*neg_b=*/false, b,
      /*c_mod=*/(short)0, c, /*reuse_a=*/false, /*reuse_b=*/false);
}

__device__ __forceinline__ float fsig(float x) {
  return 1.0f / (1.0f + __expf(-x));
}

// ---------------------------------------------------------------------------
// Init: build Wx^T, Wh^T (bf16, N-major so B fragments are contiguous),
// bf16 h0 into h double-buffer slot 0, reset grid-barrier counter.
// ---------------------------------------------------------------------------
__global__ void lstm_init_small(const float* __restrict__ Wx,
                                const float* __restrict__ Wh,
                                const float* __restrict__ h0,
                                __bf16* __restrict__ wxT,
                                __bf16* __restrict__ whT,
                                __bf16* __restrict__ hbf,
                                unsigned* __restrict__ counter) {
  int idx = blockIdx.x * blockDim.x + threadIdx.x;
  if (idx < GDIM * DDIM) {
    int n = idx / DDIM;
    int d = idx - n * DDIM;
    wxT[idx] = (__bf16)Wx[(size_t)d * GDIM + n];
    whT[idx] = (__bf16)Wh[(size_t)d * GDIM + n];
  }
  if (idx < NBATCH * HDIM) hbf[idx] = (__bf16)h0[idx];
  if (idx == 0) *counter = 0u;
}

// f32 -> bf16 copy of x (bandwidth pass; halves GEMM read traffic).
__global__ void lstm_convert_x(const float* __restrict__ x,
                               __bf16* __restrict__ xb, size_t n4) {
  size_t i = (size_t)blockIdx.x * blockDim.x + threadIdx.x;
  size_t stride = (size_t)gridDim.x * blockDim.x;
  for (; i < n4; i += stride) {
    v4f v = *(const v4f*)(x + i * 4);
    v4bf o = { (__bf16)v.x, (__bf16)v.y, (__bf16)v.z, (__bf16)v.w };
    *(v4bf*)(xb + i * 4) = o;
  }
}

// ---------------------------------------------------------------------------
// xproj GEMM: (65536 x 512) @ (512 x 2048) + b, bf16 WMMA, f32 accum.
// Block = 256 threads (8 waves, 2x4), block tile 64(M) x 128(N),
// each wave does a 32x32 tile (2x2 WMMA tiles, A/B frag reuse).
// Output rows permuted to (t*32 + n) so scan reads are contiguous per step.
// ---------------------------------------------------------------------------
__global__ __launch_bounds__(256)
void lstm_xproj_gemm(const __bf16* __restrict__ xb,
                     const __bf16* __restrict__ wxT,
                     const float* __restrict__ bias,
                     __bf16* __restrict__ xp) {
  const int lane = threadIdx.x & 31;
  const int wid  = threadIdx.x >> 5;
  const int mw = wid & 1, nw = wid >> 1;
  const int m0 = blockIdx.x * 64 + mw * 32;
  const int n0 = blockIdx.y * 128 + nw * 32;
  const int ar0 = m0 + (lane & 15), ar1 = ar0 + 16;
  const int bc0 = n0 + (lane & 15), bc1 = bc0 + 16;

  v8f acc00 = {}, acc01 = {}, acc10 = {}, acc11 = {};
  for (int k = 0; k < DDIM; k += 32) {
    v16bf a0 = load_frag(xb, ar0, DDIM, k);
    v16bf a1 = load_frag(xb, ar1, DDIM, k);
    v16bf b0 = load_frag(wxT, bc0, DDIM, k);
    v16bf b1 = load_frag(wxT, bc1, DDIM, k);
    acc00 = wmma_bf16(a0, b0, acc00);
    acc01 = wmma_bf16(a0, b1, acc01);
    acc10 = wmma_bf16(a1, b0, acc10);
    acc11 = wmma_bf16(a1, b1, acc11);
  }

  const int half = lane >> 4;
#pragma unroll
  for (int mt = 0; mt < 2; ++mt) {
#pragma unroll
    for (int nt = 0; nt < 2; ++nt) {
      v8f acc = (mt == 0) ? (nt == 0 ? acc00 : acc01)
                          : (nt == 0 ? acc10 : acc11);
      const int col = n0 + nt * 32 + (lane & 15) + 0; // nt tile base is nt*16 within wave's 32
      const int c2  = n0 + nt * 16 + (lane & 15);
      (void)col;
      const float bv = bias[c2];
#pragma unroll
      for (int r = 0; r < 8; ++r) {
        int row = m0 + mt * 16 + r + half * 8;  // global row = n*T + t
        int nb  = row >> 11;                    // / 2048
        int tt  = row & 2047;
        xp[((size_t)(tt * NBATCH + nb)) * GDIM + c2] = (__bf16)(acc[r] + bv);
      }
    }
  }
}

// ---------------------------------------------------------------------------
// Persistent recurrent scan. Grid = 64 WGs x 32 threads (1 wave each).
// WG -> (m = batch half, j0 = 16-hidden-unit slice). Per step:
//   act = xp[t] + h_{t-1} @ Wh  via 64 bf16 WMMAs (A from global h, B from LDS)
//   gate fusion fully register-local (i,f,o,g tiles share the C layout),
//   c kept in 8 VGPRs across all 2048 steps, h double-buffered in bf16.
// Device-wide sync: monotonically increasing L2 atomic counter.
// ---------------------------------------------------------------------------
__global__ __launch_bounds__(32)
void lstm_scan(const __bf16* __restrict__ xp,
               const __bf16* __restrict__ whT,
               __bf16* __restrict__ hbuf,            // 2 buffers of 32x512
               float* __restrict__ out,
               unsigned* __restrict__ counter) {
  __shared__ __bf16 whs[64 * 32 * 16];               // 64 KB: 4 gates x 16 k-steps

  const int lane = threadIdx.x & 31;
  const int half = lane >> 4;
  const int m  = blockIdx.x & 1;
  const int j0 = (blockIdx.x >> 1) * 16;
  const unsigned nwg = gridDim.x;

  // Preload this WG's Wh^T slice into LDS in exact fragment layout.
  for (int slot = 0; slot < 64; ++slot) {
    int g = slot >> 4, kk = slot & 15;
    int col = g * HDIM + j0 + (lane & 15);
    const __bf16* p = whT + (size_t)col * DDIM + kk * 32 + half * 8;
    v8bf lo = *(const v8bf*)p;
    v8bf hi = *(const v8bf*)(p + 16);
    __bf16* q = &whs[(slot * 32 + lane) * 16];
    *(v8bf*)q = lo;
    *(v8bf*)(q + 8) = hi;
  }
  __syncthreads();

  v8f c = {};
  const int arow  = m * 16 + (lane & 15);
  const int nbase = m * 16 + half * 8;
  const int col   = j0 + (lane & 15);

  for (int t = 0; t < TSTEPS; ++t) {
    const __bf16* hread  = hbuf + (size_t)(t & 1) * (NBATCH * HDIM);
    __bf16*       hwrite = hbuf + (size_t)((t + 1) & 1) * (NBATCH * HDIM);

    if (t + 1 < TSTEPS)   // warm L2 for next step's activations
      __builtin_prefetch(xp + ((size_t)(t + 1) * NBATCH + nbase) * GDIM + col, 0, 0);

    v8f a0 = {}, a1 = {}, a2 = {}, a3 = {};
#pragma unroll 4
    for (int kk = 0; kk < 16; ++kk) {
      v16bf a  = load_frag(hread, arow, HDIM, kk * 32);
      v16bf b0 = *(const v16bf*)&whs[(( 0 + kk) * 32 + lane) * 16];
      v16bf b1 = *(const v16bf*)&whs[((16 + kk) * 32 + lane) * 16];
      v16bf b2 = *(const v16bf*)&whs[((32 + kk) * 32 + lane) * 16];
      v16bf b3 = *(const v16bf*)&whs[((48 + kk) * 32 + lane) * 16];
      a0 = wmma_bf16(a, b0, a0);
      a1 = wmma_bf16(a, b1, a1);
      a2 = wmma_bf16(a, b2, a2);
      a3 = wmma_bf16(a, b3, a3);
    }

    const __bf16* xrow = xp + (size_t)t * NBATCH * GDIM;
#pragma unroll
    for (int r = 0; r < 8; ++r) {
      int nb = nbase + r;
      const __bf16* xr = xrow + (size_t)nb * GDIM;
      float ai = a0[r] + (float)xr[0 * HDIM + col];
      float af = a1[r] + (float)xr[1 * HDIM + col];
      float ao = a2[r] + (float)xr[2 * HDIM + col];
      float ag = a3[r] + (float)xr[3 * HDIM + col];
      float iv = fsig(ai), fv = fsig(af), ov = fsig(ao);
      float gv = tanhf(ag);
      float cv = fv * c[r] + iv * gv;
      c[r] = cv;
      float hv = ov * tanhf(cv);
      hwrite[(size_t)nb * HDIM + col] = (__bf16)hv;
      out[((size_t)nb * TSTEPS + t) * HDIM + col] = hv;
    }

    // ---- device-wide sense barrier (monotonic counter, reset at launch) ----
    __threadfence();
    if (lane == 0) {
      __hip_atomic_fetch_add(counter, 1u, __ATOMIC_ACQ_REL, __HIP_MEMORY_SCOPE_AGENT);
      const unsigned target = (unsigned)(t + 1) * nwg;
      while (__hip_atomic_load(counter, __ATOMIC_ACQUIRE, __HIP_MEMORY_SCOPE_AGENT) < target)
        __builtin_amdgcn_s_sleep(1);
    }
    __threadfence();
  }
}

// ---------------------------------------------------------------------------
extern "C" void kernel_launch(void* const* d_in, const int* in_sizes, int n_in,
                              void* d_out, int out_size, void* d_ws, size_t ws_size,
                              hipStream_t stream) {
  const float* x  = (const float*)d_in[0];   // (32, 2048, 512)
  const float* h0 = (const float*)d_in[1];   // (32, 512)
  const float* Wx = (const float*)d_in[2];   // (512, 2048)
  const float* Wh = (const float*)d_in[3];   // (512, 2048)
  const float* b  = (const float*)d_in[4];   // (2048,)
  float* out = (float*)d_out;                // (32, 2048, 512)

  char* ws = (char*)d_ws;
  unsigned* counter = (unsigned*)(ws + OFF_CNT);
  __bf16* hbf = (__bf16*)(ws + OFF_HBF);
  __bf16* wxT = (__bf16*)(ws + OFF_WXT);
  __bf16* whT = (__bf16*)(ws + OFF_WHT);
  __bf16* xb  = (__bf16*)(ws + OFF_XB);
  __bf16* xp  = (__bf16*)(ws + OFF_XP);

  (void)in_sizes; (void)n_in; (void)out_size; (void)ws_size;

  lstm_init_small<<<(GDIM * DDIM + 255) / 256, 256, 0, stream>>>(
      Wx, Wh, h0, wxT, whT, hbf, counter);

  lstm_convert_x<<<4096, 256, 0, stream>>>(
      x, xb, (size_t)NBATCH * TSTEPS * DDIM / 4);

  dim3 gg(65536 / 64, GDIM / 128);
  lstm_xproj_gemm<<<gg, 256, 0, stream>>>(xb, wxT, b, xp);

  lstm_scan<<<64, 32, 0, stream>>>(xp, whT, hbf, out, counter);
}
